// Raindrop2_24910810316830
// MI455X (gfx1250) — matrix operand order repaired and checked by reading, not compiled, not run on hardware
//
#include <hip/hip_runtime.h>
#include <math.h>

typedef _Float16 half_t;
typedef __attribute__((ext_vector_type(16))) _Float16 v16h;
typedef __attribute__((ext_vector_type(8)))  _Float16 v8h;
typedef __attribute__((ext_vector_type(8)))  float    v8f;

#define T_N   500
#define B_N   128
#define S_N   36
#define D_N   500
#define DENC  72
#define DST   9
#define DMODEL 64
#define NCLS  3
#define KPG   512   // padded K / N for graph GEMMs (500 -> 512)
#define SP    48    // padded sensor rows (36 -> 48, 3 M-tiles)
#define KPE   96    // padded encoder width (72 -> 96)
#define NPQKV 224   // padded 3*72 = 216 -> 224
#define TP    512   // padded time (500 -> 512)
#define MTOK  (B_N * T_N)   // 64000 tokens (exactly 4000 M-tiles)
#define CHUNK 16    // batches per attention chunk

// ---------------------------------------------------------------------------
// WMMA helpers: A row-major f16 (hopscotch fragment), B given as Bt (N x K
// row-major, read with the same hopscotch fragment as A), C f32 row-major.
// ---------------------------------------------------------------------------
__device__ inline v16h frag_a_f16(const half_t* __restrict__ ap) {
  // ap points at row base + kb + 8*h ; elements e<8 -> k=e, e>=8 -> k=e+8
  v8h a0 = *(const v8h*)(ap);
  v8h a1 = *(const v8h*)(ap + 16);
  return __builtin_shufflevector(a0, a1, 0,1,2,3,4,5,6,7,8,9,10,11,12,13,14,15);
}
__device__ inline v16h frag_b_f16(const half_t* __restrict__ bp) {
  // bp points at Bt row base + kb + 8*h ; elements e<8 -> k=e, e>=8 -> k=e+8
  v8h b0 = *(const v8h*)(bp);
  v8h b1 = *(const v8h*)(bp + 16);
  return __builtin_shufflevector(b0, b1, 0,1,2,3,4,5,6,7,8,9,10,11,12,13,14,15);
}
__device__ inline v8f wmma32(v16h a, v16h b, v8f c) {
  return __builtin_amdgcn_wmma_f32_16x16x32_f16(false, a, false, b, (short)0, c,
                                                false, false);
}

// ---------------------------------------------------------------------------
// Register-blocked batched GEMM: each wave computes MB vertically-adjacent
// 16x16 tiles of C[bz] = scale * (A[bz] @ Bt[bz]^T), sharing one B-fragment
// per K-step across MB WMMAs (B traffic / MB).
// ---------------------------------------------------------------------------
template <int MB>
__global__ void k_gemm_f16_t(const half_t* __restrict__ A, int lda, long sA,
                             const half_t* __restrict__ Bt, int ldb, long sB,
                             float* __restrict__ C, int ldc, long sC,
                             int K, float scale) {
  const int lane = threadIdx.x;
  const int nt = blockIdx.x, mt0 = blockIdx.y * MB, bz = blockIdx.z;
  const int r = lane & 15, h = lane >> 4;
  const half_t* Ab = A + (long)bz * sA + (long)(mt0 * 16 + r) * lda;
  const half_t* Bb = Bt + (long)bz * sB + (long)(nt * 16 + r) * ldb;
  v8f acc[MB];
#pragma unroll
  for (int j = 0; j < MB; ++j) acc[j] = {};
  for (int kb = 0; kb < K; kb += 32) {
    v16h bv = frag_b_f16(Bb + kb + 8 * h);
#pragma unroll
    for (int j = 0; j < MB; ++j) {
      v16h av = frag_a_f16(Ab + (long)j * 16 * lda + kb + 8 * h);
      acc[j] = wmma32(av, bv, acc[j]);
    }
  }
  float* Cb = C + (long)bz * sC + (nt * 16 + r);
#pragma unroll
  for (int j = 0; j < MB; ++j) {
    const int mbase = (mt0 + j) * 16 + 8 * h;
#pragma unroll
    for (int i = 0; i < 8; ++i)
      Cb[(long)(mbase + i) * ldc] = acc[j][i] * scale;
  }
}

// Same but A is f32 (attention probabilities), converted in-register to f16.
template <int MB>
__global__ void k_gemm_f32a_t(const float* __restrict__ A, int lda, long sA,
                              const half_t* __restrict__ Bt, int ldb, long sB,
                              float* __restrict__ C, int ldc, long sC, int K) {
  const int lane = threadIdx.x;
  const int nt = blockIdx.x, mt0 = blockIdx.y * MB, bz = blockIdx.z;
  const int r = lane & 15, h = lane >> 4;
  const float* Ab = A + (long)bz * sA + (long)(mt0 * 16 + r) * lda;
  const half_t* Bb = Bt + (long)bz * sB + (long)(nt * 16 + r) * ldb;
  v8f acc[MB];
#pragma unroll
  for (int j = 0; j < MB; ++j) acc[j] = {};
  for (int kb = 0; kb < K; kb += 32) {
    v16h bv = frag_b_f16(Bb + kb + 8 * h);
#pragma unroll
    for (int j = 0; j < MB; ++j) {
      const float* ap = Ab + (long)j * 16 * lda + kb + 8 * h;
      v8f f0 = *(const v8f*)(ap);
      v8f f1 = *(const v8f*)(ap + 16);
      v16h av;
#pragma unroll
      for (int i = 0; i < 8; ++i) {
        av[i] = (_Float16)f0[i];
        av[i + 8] = (_Float16)f1[i];
      }
      acc[j] = wmma32(av, bv, acc[j]);
    }
  }
  float* Cb = C + (long)bz * sC + (nt * 16 + r);
#pragma unroll
  for (int j = 0; j < MB; ++j) {
    const int mbase = (mt0 + j) * 16 + 8 * h;
#pragma unroll
    for (int i = 0; i < 8; ++i)
      Cb[(long)(mbase + i) * ldc] = acc[j][i];
  }
}

// Transformer QKV GEMM (M-blocked x4) with fused bias + scatter into Q16/K16
// (row-major, A/Bt operand layout for scores) and Vt16 (transposed, Bt layout
// for PV).
__global__ void k_gemm_qkvtr(const half_t* __restrict__ x16,
                             const half_t* __restrict__ Wt,
                             const float* __restrict__ bqkv,
                             half_t* __restrict__ Q16,
                             half_t* __restrict__ K16,
                             half_t* __restrict__ Vt16) {
  constexpr int MB = 4;
  const int lane = threadIdx.x;
  const int nt = blockIdx.x, mt0 = blockIdx.y * MB;
  const int r = lane & 15, h = lane >> 4;
  const half_t* Ab = x16 + (long)(mt0 * 16 + r) * KPE;
  const half_t* Bb = Wt + (long)(nt * 16 + r) * KPE;
  v8f acc[MB];
#pragma unroll
  for (int j = 0; j < MB; ++j) acc[j] = {};
#pragma unroll
  for (int kb = 0; kb < KPE; kb += 32) {
    v16h bv = frag_b_f16(Bb + kb + 8 * h);
#pragma unroll
    for (int j = 0; j < MB; ++j) {
      v16h av = frag_a_f16(Ab + (long)j * 16 * KPE + kb + 8 * h);
      acc[j] = wmma32(av, bv, acc[j]);
    }
  }
  const int n = nt * 16 + r;
  const float bias = (n < 3 * DENC) ? bqkv[n] : 0.f;
#pragma unroll
  for (int j = 0; j < MB; ++j) {
#pragma unroll
    for (int i = 0; i < 8; ++i) {
      const int m = (mt0 + j) * 16 + 8 * h + i;  // global token, 0..63999
      const int b = m / T_N, t = m % T_N;
      const float val = acc[j][i] + bias;
      if (n < DENC)
        Q16[((long)b * TP + t) * KPE + n] = (_Float16)val;
      else if (n < 2 * DENC)
        K16[((long)b * TP + t) * KPE + (n - DENC)] = (_Float16)val;
      else if (n < 3 * DENC)
        Vt16[((long)b * KPE + (n - 2 * DENC)) * TP + t] = (_Float16)val;
    }
  }
}

// ---------------------------------------------------------------------------
// Data movement / small kernels
// ---------------------------------------------------------------------------
__global__ void k_build_xg16(const float* __restrict__ src,
                             half_t* __restrict__ Xg16) {
  const long id = (long)blockIdx.x * 256 + threadIdx.x;  // B*SP*KPG
  const int b = id / (SP * KPG);
  const int rem = id % (SP * KPG);
  const int s = rem / KPG, k = rem % KPG;
  _Float16 v = (_Float16)0.f;
  if (s < S_N && k < D_N)
    v = (_Float16)src[(long)k * B_N * S_N + b * S_N + s];  // src[T,B,S]
  Xg16[id] = v;
}

__global__ void k_convW_graph(const float* __restrict__ W,  // [500][500]
                              half_t* __restrict__ Wt) {    // [512][512] = W^T
  const long id = (long)blockIdx.x * 256 + threadIdx.x;
  const int n = id / KPG, k = id % KPG;
  Wt[id] = (n < D_N && k < D_N) ? (_Float16)W[(long)k * D_N + n]
                                : (_Float16)0.f;
}

__global__ void k_convW_qkv(const float* __restrict__ W,  // [72][216]
                            half_t* __restrict__ Wt) {    // [224][96] = W^T
  const long id = (long)blockIdx.x * 256 + threadIdx.x;
  const int n = id / KPE, k = id % KPE;
  Wt[id] = (n < 3 * DENC && k < DENC) ? (_Float16)W[(long)k * 3 * DENC + n]
                                      : (_Float16)0.f;
}

// Graph attention logits (36x36 per batch, K=500) + edge-softmax.
__global__ void k_graph_attn(const float* __restrict__ gQ,
                             const float* __restrict__ gK,
                             const float* __restrict__ ew,  // nullptr layer 0
                             float* __restrict__ alphaOut) {
  const int b = blockIdx.x, n = blockIdx.y, m = threadIdx.x;  // block 64
  __shared__ float lg[S_N];
  __shared__ float mx, sm;
  if (m < S_N) {
    const float* q = gQ + ((long)b * SP + n) * KPG;
    const float* k = gK + ((long)b * SP + m) * KPG;
    float s = 0.f;
    for (int d = 0; d < D_N; ++d) s += q[d] * k[d];
    s *= 0.04472135955f;  // 1/sqrt(500)
    if (ew) s *= ew[((long)b * S_N + n) * S_N + m];
    lg[m] = s;
  }
  __syncthreads();
  if (threadIdx.x == 0) {
    float M = lg[0];
    for (int i = 1; i < S_N; ++i) M = fmaxf(M, lg[i]);
    float S2 = 0.f;
    for (int i = 0; i < S_N; ++i) S2 += expf(lg[i] - M);
    mx = M;
    sm = S2;
  }
  __syncthreads();
  if (m < S_N)
    alphaOut[((long)b * S_N + n) * S_N + m] = expf(lg[m] - mx) / sm;
}

// xg_new = alpha @ v ; also refresh f16 operand for next layer.
__global__ void k_graph_av(const float* __restrict__ alpha,
                           const float* __restrict__ gV,
                           float* __restrict__ xg,
                           half_t* __restrict__ Xg16) {
  const int b = blockIdx.x, n = blockIdx.y, e = threadIdx.x;  // block 512
  __shared__ float a[S_N];
  if (e < S_N) a[e] = alpha[((long)b * S_N + n) * S_N + e];
  __syncthreads();
  float s = 0.f;
  for (int m = 0; m < S_N; ++m) s += a[m] * gV[((long)b * SP + m) * KPG + e];
  xg[((long)b * S_N + n) * KPG + e] = s;
  Xg16[((long)b * SP + n) * KPG + e] = (_Float16)s;
}

__global__ void k_distance(const float* __restrict__ alpha,
                           float* __restrict__ acc) {
  const int i = blockIdx.x, j = threadIdx.x;  // 128 x 128
  const float* ai = alpha + (long)i * (S_N * S_N);
  const float* aj = alpha + (long)j * (S_N * S_N);
  float sii = 0.f, sjj = 0.f, sij = 0.f;
  for (int k = 0; k < S_N * S_N; ++k) {
    const float a = ai[k], b = aj[k];
    sii += a * a;
    sjj += b * b;
    sij += a * b;
  }
  const float d2 = sii + sjj - 2.f * sij;
  atomicAdd(acc, sqrtf(fmaxf(d2, 1e-12f)));
}

// Build transformer input x[b,t,f] (f<36: graph out, f>=36: Raindrop PE) and
// its padded f16 operand copy.
__global__ void k_build_x(const float* __restrict__ xg,
                          const float* __restrict__ times,
                          float* __restrict__ x, half_t* __restrict__ x16) {
  const long id = (long)blockIdx.x * 256 + threadIdx.x;  // B*T*KPE
  const int b = id / (T_N * KPE);
  const int rem = id % (T_N * KPE);
  const int t = rem / KPE, f = rem % KPE;
  float v = 0.f;
  if (f < S_N) {
    v = xg[((long)b * S_N + f) * KPG + t];
  } else if (f < DENC) {
    const int j = f - S_N;
    const int jj = (j < 18) ? j : j - 18;
    const float ts = powf(500.f, (float)jj / 18.f);
    const float sc = times[(long)b * T_N + t] / ts;
    v = (j < 18) ? sinf(sc) : cosf(sc);
  }
  if (f < DENC) x[((long)b * T_N + t) * DENC + f] = v;
  x16[((long)b * T_N + t) * KPE + f] = (_Float16)v;
}

// Row softmax over 500 keys with key-padding mask; zero the 12 pad columns.
__global__ void k_softmax(float* __restrict__ scores,
                          const int* __restrict__ lengths, int chunkStart) {
  const int bl = blockIdx.x / T_N, q = blockIdx.x % T_N;
  const int len = lengths[chunkStart + bl];
  float* p = scores + ((long)bl * TP + q) * TP;
  const int k = threadIdx.x;  // 512
  float v = -3.0e38f;
  if (k < T_N) {
    v = p[k];
    if (k >= len) v += -1e9f;
  }
  __shared__ float red[TP];
  red[k] = v;
  __syncthreads();
  for (int off = 256; off > 0; off >>= 1) {
    if (k < off) red[k] = fmaxf(red[k], red[k + off]);
    __syncthreads();
  }
  const float mx = red[0];
  __syncthreads();
  const float e = (k < T_N) ? expf(v - mx) : 0.f;
  red[k] = e;
  __syncthreads();
  for (int off = 256; off > 0; off >>= 1) {
    if (k < off) red[k] += red[k + off];
    __syncthreads();
  }
  p[k] = e / red[0];
}

// Per-token: attn @ Wo + bo, residual, LN1, FFN (d_ff=1), residual, LN2.
__global__ void k_proj_ln(const float* __restrict__ attn,
                          const float* __restrict__ Wo,
                          const float* __restrict__ bo,
                          const float* __restrict__ ln1g,
                          const float* __restrict__ ln1b,
                          const float* __restrict__ ffW1,
                          const float* __restrict__ ffb1,
                          const float* __restrict__ ffW2,
                          const float* __restrict__ ffb2,
                          const float* __restrict__ ln2g,
                          const float* __restrict__ ln2b,
                          float* __restrict__ x, half_t* __restrict__ x16) {
  const long tok = blockIdx.x;  // 64000
  const int b = tok / T_N, t = tok % T_N;
  const int f = threadIdx.x;  // 96
  __shared__ float o[DENC], xr[DENC];
  __shared__ float mu, rstd, hid;
  if (f < DENC) o[f] = attn[((long)b * TP + t) * KPE + f];
  __syncthreads();
  if (f < DENC) {
    float s = bo[f];
    for (int g = 0; g < DENC; ++g) s += o[g] * Wo[g * DENC + f];
    xr[f] = x[tok * DENC + f] + s;
  }
  __syncthreads();
  if (f == 0) {
    float m = 0.f;
    for (int i = 0; i < DENC; ++i) m += xr[i];
    m /= DENC;
    float v = 0.f;
    for (int i = 0; i < DENC; ++i) {
      const float d = xr[i] - m;
      v += d * d;
    }
    mu = m;
    rstd = rsqrtf(v / DENC + 1e-5f);
  }
  __syncthreads();
  if (f < DENC) xr[f] = (xr[f] - mu) * rstd * ln1g[f] + ln1b[f];
  __syncthreads();
  if (f == 0) {
    float s = ffb1[0];
    for (int i = 0; i < DENC; ++i) s += xr[i] * ffW1[i];
    hid = fmaxf(s, 0.f);
  }
  __syncthreads();
  if (f < DENC) xr[f] = xr[f] + hid * ffW2[f] + ffb2[f];
  __syncthreads();
  if (f == 0) {
    float m = 0.f;
    for (int i = 0; i < DENC; ++i) m += xr[i];
    m /= DENC;
    float v = 0.f;
    for (int i = 0; i < DENC; ++i) {
      const float d = xr[i] - m;
      v += d * d;
    }
    mu = m;
    rstd = rsqrtf(v / DENC + 1e-5f);
  }
  __syncthreads();
  if (f < DENC) {
    const float y = (xr[f] - mu) * rstd * ln2g[f] + ln2b[f];
    x[tok * DENC + f] = y;
    x16[tok * KPE + f] = (_Float16)y;
  }
}

// Masked mean over time, static embed, 81->64->3 MLP, distance write-out.
__global__ void k_final(const float* __restrict__ x,
                        const int* __restrict__ lengths,
                        const float* __restrict__ statics,
                        const float* __restrict__ statW,
                        const float* __restrict__ statb,
                        const float* __restrict__ W1,
                        const float* __restrict__ b1,
                        const float* __restrict__ W2,
                        const float* __restrict__ b2,
                        const float* __restrict__ distAcc,
                        float* __restrict__ out) {
  const int b = blockIdx.x;
  const int tid = threadIdx.x;  // 128
  __shared__ float feat[DENC + DST], hid[DMODEL];
  const int len = lengths[b];
  if (tid < DENC) {
    float s = 0.f;
    for (int t = 0; t < T_N; ++t)
      if (t < len) s += x[((long)b * T_N + t) * DENC + tid];
    feat[tid] = s / (float)(len + 1);
  } else if (tid < DENC + DST) {
    const int j = tid - DENC;
    float s = statb[j];
    for (int i = 0; i < DST; ++i) s += statics[b * DST + i] * statW[i * DST + j];
    feat[tid] = s;
  }
  __syncthreads();
  if (tid < DMODEL) {
    float s = b1[tid];
    for (int i = 0; i < DENC + DST; ++i) s += feat[i] * W1[i * DMODEL + tid];
    hid[tid] = fmaxf(s, 0.f);
  }
  __syncthreads();
  if (tid < NCLS) {
    float s = b2[tid];
    for (int j = 0; j < DMODEL; ++j) s += hid[j] * W2[j * NCLS + tid];
    out[b * NCLS + tid] = s;
  }
  if (b == 0 && tid == 0)
    out[B_N * NCLS] = distAcc[0] * (1.f / ((float)B_N * (float)B_N));
}

// ---------------------------------------------------------------------------
extern "C" void kernel_launch(void* const* d_in, const int* in_sizes, int n_in,
                              void* d_out, int out_size, void* d_ws,
                              size_t ws_size, hipStream_t stream) {
  (void)in_sizes; (void)n_in; (void)out_size; (void)ws_size;
  const float* src     = (const float*)d_in[0];
  const float* statics = (const float*)d_in[1];
  const float* times   = (const float*)d_in[2];
  const int*   lengths = (const int*)d_in[3];
  const float* op_Wq   = (const float*)d_in[6];
  const float* op_Wk   = (const float*)d_in[7];
  const float* op_Wv   = (const float*)d_in[8];
  const float* stat_W  = (const float*)d_in[9];
  const float* stat_b  = (const float*)d_in[10];
  const float* te_Wqkv = (const float*)d_in[11];
  const float* te_bqkv = (const float*)d_in[12];
  const float* te_Wo   = (const float*)d_in[13];
  const float* te_bo   = (const float*)d_in[14];
  const float* te_ln1g = (const float*)d_in[15];
  const float* te_ln1b = (const float*)d_in[16];
  const float* te_ffW1 = (const float*)d_in[17];
  const float* te_ffb1 = (const float*)d_in[18];
  const float* te_ffW2 = (const float*)d_in[19];
  const float* te_ffb2 = (const float*)d_in[20];
  const float* te_ln2g = (const float*)d_in[21];
  const float* te_ln2b = (const float*)d_in[22];
  const float* mlp_W1  = (const float*)d_in[23];
  const float* mlp_b1  = (const float*)d_in[24];
  const float* mlp_W2  = (const float*)d_in[25];
  const float* mlp_b2  = (const float*)d_in[26];

  char* wp = (char*)d_ws;
  auto alloc = [&](size_t bytes) {
    void* p = (void*)wp;
    wp += (bytes + 255) & ~(size_t)255;
    return p;
  };
  half_t* Xg16 = (half_t*)alloc((size_t)B_N * SP * KPG * 2);
  half_t* WtQ  = (half_t*)alloc((size_t)KPG * KPG * 2);
  half_t* WtK  = (half_t*)alloc((size_t)KPG * KPG * 2);
  half_t* WtV  = (half_t*)alloc((size_t)KPG * KPG * 2);
  float*  gQ   = (float*)alloc((size_t)B_N * SP * KPG * 4);
  float*  gK   = (float*)alloc((size_t)B_N * SP * KPG * 4);
  float*  gV   = (float*)alloc((size_t)B_N * SP * KPG * 4);
  float*  alphaA = (float*)alloc((size_t)B_N * S_N * S_N * 4);
  float*  alphaB = (float*)alloc((size_t)B_N * S_N * S_N * 4);
  float*  xg   = (float*)alloc((size_t)B_N * S_N * KPG * 4);
  float*  distAcc = (float*)alloc(256);
  float*  x    = (float*)alloc((size_t)B_N * T_N * DENC * 4);
  half_t* x16  = (half_t*)alloc((size_t)MTOK * KPE * 2);
  half_t* Wqkvt = (half_t*)alloc((size_t)NPQKV * KPE * 2);
  half_t* Q16  = (half_t*)alloc((size_t)B_N * TP * KPE * 2);
  half_t* K16  = (half_t*)alloc((size_t)B_N * TP * KPE * 2);
  half_t* Vt16 = (half_t*)alloc((size_t)B_N * KPE * TP * 2);
  float*  scores = (float*)alloc((size_t)CHUNK * TP * TP * 4);
  float*  attn = (float*)alloc((size_t)B_N * TP * KPE * 4);

  // ---- graph attention over sensors ----
  k_build_xg16<<<(B_N * SP * KPG) / 256, 256, 0, stream>>>(src, Xg16);
  for (int l = 0; l < 2; ++l) {
    const long wo = (long)l * D_N * D_N;
    k_convW_graph<<<(KPG * KPG) / 256, 256, 0, stream>>>(op_Wq + wo, WtQ);
    k_convW_graph<<<(KPG * KPG) / 256, 256, 0, stream>>>(op_Wk + wo, WtK);
    k_convW_graph<<<(KPG * KPG) / 256, 256, 0, stream>>>(op_Wv + wo, WtV);
    // M = 48 rows = 3 tiles, all in one wave (MB=3): weight matrix read once
    // per batch.
    k_gemm_f16_t<3><<<dim3(KPG / 16, 1, B_N), 32, 0, stream>>>(
        Xg16, KPG, (long)SP * KPG, WtQ, KPG, 0, gQ, KPG, (long)SP * KPG, KPG,
        1.f);
    k_gemm_f16_t<3><<<dim3(KPG / 16, 1, B_N), 32, 0, stream>>>(
        Xg16, KPG, (long)SP * KPG, WtK, KPG, 0, gK, KPG, (long)SP * KPG, KPG,
        1.f);
    k_gemm_f16_t<3><<<dim3(KPG / 16, 1, B_N), 32, 0, stream>>>(
        Xg16, KPG, (long)SP * KPG, WtV, KPG, 0, gV, KPG, (long)SP * KPG, KPG,
        1.f);
    const float* ew = (l == 0) ? (const float*)nullptr : alphaA;
    float* aOut = (l == 0) ? alphaA : alphaB;
    k_graph_attn<<<dim3(B_N, S_N), 64, 0, stream>>>(gQ, gK, ew, aOut);
    k_graph_av<<<dim3(B_N, S_N), KPG, 0, stream>>>(aOut, gV, xg, Xg16);
  }
  hipMemsetAsync(distAcc, 0, 4, stream);
  k_distance<<<B_N, B_N, 0, stream>>>(alphaB, distAcc);

  // ---- transformer encoder over time ----
  k_build_x<<<(B_N * T_N * KPE) / 256, 256, 0, stream>>>(xg, times, x, x16);
  for (int l = 0; l < 2; ++l) {
    k_convW_qkv<<<(NPQKV * KPE) / 256, 256, 0, stream>>>(
        te_Wqkv + (long)l * DENC * 3 * DENC, Wqkvt);
    hipMemsetAsync(Q16, 0, (size_t)B_N * TP * KPE * 2, stream);
    hipMemsetAsync(K16, 0, (size_t)B_N * TP * KPE * 2, stream);
    hipMemsetAsync(Vt16, 0, (size_t)B_N * KPE * TP * 2, stream);
    k_gemm_qkvtr<<<dim3(NPQKV / 16, MTOK / 16 / 4), 32, 0, stream>>>(
        x16, Wqkvt, te_bqkv + (long)l * 3 * DENC, Q16, K16, Vt16);
    for (int c = 0; c < B_N / CHUNK; ++c) {
      const long cs = (long)c * CHUNK;
      // scores = Q @ K^T / sqrt(72): MB=4 -> K-operand traffic / 4
      k_gemm_f16_t<4><<<dim3(TP / 16, TP / 16 / 4, CHUNK), 32, 0, stream>>>(
          Q16 + cs * TP * KPE, KPE, (long)TP * KPE,
          K16 + cs * TP * KPE, KPE, (long)TP * KPE,
          scores, TP, (long)TP * TP, KPE, 0.11785113f /* 1/sqrt(72) */);
      k_softmax<<<CHUNK * T_N, TP, 0, stream>>>(scores, lengths, (int)cs);
      // O = P @ V: MB=4 -> V-operand traffic / 4
      k_gemm_f32a_t<4><<<dim3(KPE / 16, TP / 16 / 4, CHUNK), 32, 0, stream>>>(
          scores, TP, (long)TP * TP,
          Vt16 + cs * KPE * TP, TP, (long)KPE * TP,
          attn + cs * TP * KPE, KPE, (long)TP * KPE, TP);
    }
    k_proj_ln<<<MTOK, KPE, 0, stream>>>(
        attn, te_Wo + (long)l * DENC * DENC, te_bo + (long)l * DENC,
        te_ln1g + (long)l * DENC, te_ln1b + (long)l * DENC,
        te_ffW1 + (long)l * DENC, te_ffb1 + l, te_ffW2 + (long)l * DENC,
        te_ffb2 + (long)l * DENC, te_ln2g + (long)l * DENC,
        te_ln2b + (long)l * DENC, x, x16);
  }

  // ---- aggregation + classifier head (+ distance scalar) ----
  k_final<<<B_N, 128, 0, stream>>>(x, lengths, statics, stat_W, stat_b, mlp_W1,
                                   mlp_b1, mlp_W2, mlp_b2, distAcc,
                                   (float*)d_out);
}
